// AFTConv_76218489635105
// MI455X (gfx1250) — compile-verified, matrix operand order checked by reading, not run on hardware
//
#include <hip/hip_runtime.h>
#include <hip/hip_bf16.h>
#include <math.h>

// ---------------------------------------------------------------------------
// AFT-Conv on MI455X (gfx1250): bf16 WMMA for the three 512x512 projections
// with double-buffered GLOBAL_LOAD_ASYNC_TO_LDS (ASYNCcnt) staging; fused
// VALU conv/combine for the cheap depthwise part.
// ---------------------------------------------------------------------------

typedef __bf16 bf16_t;
typedef __attribute__((ext_vector_type(16))) __bf16 v16bf;
typedef __attribute__((ext_vector_type(8)))  __bf16 bf16x8;
typedef __attribute__((ext_vector_type(8)))  float  v8f;

static constexpr int BATCH = 16;
static constexpr int HH    = 48;
static constexpr int WW    = 48;
static constexpr int DIM   = 512;
static constexpr int HEADS = 8;
static constexpr int DH    = 64;
static constexpr int NTOK  = HH * WW;            // 2304
static constexpr int ROWS  = BATCH * NTOK;       // 36864
static constexpr float EPSV = 1e-8f;

// ------------------------ CDNA5 async-copy helpers -------------------------

__device__ __forceinline__ void async_load_b128(unsigned int lds_off,
                                                const void* gaddr) {
  // GLOBAL_LOAD_ASYNC_TO_LDS_B128: vdst = LDS byte address, vaddr = 64-bit
  // global address, saddr = off (null). Tracked by ASYNCcnt.
  asm volatile("global_load_async_to_lds_b128 %0, %1, off"
               :
               : "v"(lds_off), "v"(gaddr)
               : "memory");
}

__device__ __forceinline__ void wait_async_le4() {
#if __has_builtin(__builtin_amdgcn_s_wait_asynccnt)
  __builtin_amdgcn_s_wait_asynccnt(4);
#else
  asm volatile("s_wait_asynccnt 0x4" ::: "memory");
#endif
}

__device__ __forceinline__ void wait_async_0() {
#if __has_builtin(__builtin_amdgcn_s_wait_asynccnt)
  __builtin_amdgcn_s_wait_asynccnt(0);
#else
  asm volatile("s_wait_asynccnt 0x0" ::: "memory");
#endif
}

// -------------------------- elementwise converts ---------------------------

__global__ __launch_bounds__(256) void convert_bf16(const float* __restrict__ src,
                                                    bf16_t* __restrict__ dst,
                                                    long total) {
  long i = (long)blockIdx.x * blockDim.x + threadIdx.x;
  long stride = (long)gridDim.x * blockDim.x;
  for (; i < total; i += stride) dst[i] = (bf16_t)src[i];
}

// dst (N x K) = transpose of src (K x N), converted to bf16
__global__ __launch_bounds__(256) void transpose_to_bf16(const float* __restrict__ src,
                                                         bf16_t* __restrict__ dst,
                                                         int K, int N) {
  long total = (long)K * N;
  long i = (long)blockIdx.x * blockDim.x + threadIdx.x;
  long stride = (long)gridDim.x * blockDim.x;
  for (; i < total; i += stride) {
    int n = (int)(i / K);
    int k = (int)(i % K);
    dst[i] = (bf16_t)src[(long)k * N + n];
  }
}

// ---------------------- BatchNorm(w) -> exp(wn) - 1 ------------------------

__global__ __launch_bounds__(128) void wexp_kernel(const float* __restrict__ w,
                                                   const float* __restrict__ gamma,
                                                   const float* __restrict__ beta,
                                                   float* __restrict__ wexp) {
  const int NW = HEADS * 49;  // 392
  __shared__ float s1[128];
  __shared__ float s2[128];
  int t = threadIdx.x;
  float sum = 0.f, sq = 0.f;
  for (int i = t; i < NW; i += 128) { float v = w[i]; sum += v; sq += v * v; }
  s1[t] = sum; s2[t] = sq;
  __syncthreads();
  for (int off = 64; off > 0; off >>= 1) {
    if (t < off) { s1[t] += s1[t + off]; s2[t] += s2[t + off]; }
    __syncthreads();
  }
  float mu  = s1[0] / (float)NW;
  float var = s2[0] / (float)NW - mu * mu;
  float rs  = rsqrtf(var + EPSV);
  float g = gamma[0], be = beta[0];
  for (int i = t; i < NW; i += 128)
    wexp[i] = expf(g * (w[i] - mu) * rs + be) - 1.f;
}

// ------------------------------ WMMA GEMM ----------------------------------
// C[M,N] = act(A[M,K] @ BT[N,K]^T)   A,BT bf16, C f32.
// actMode: 0 = none, 1 = sigmoid, 2 = +bias
// Block tile 128x128, BK=32, double-buffered async LDS staging with a peeled
// final iteration so the steady-state loop is branch-free.
// 8 waves (4 row-groups x 2 col-groups); each wave owns 32x64 = 2x4 tiles.

static constexpr int BM = 128;
static constexpr int BN = 128;
static constexpr int BK = 32;
static constexpr int LDP = BK + 8;  // padded LDS row (bf16 elems), 80B stride

__global__ __launch_bounds__(256) void gemm_bf16_wmma(const bf16_t* __restrict__ A,
                                                      const bf16_t* __restrict__ BT,
                                                      float* __restrict__ C,
                                                      const float* __restrict__ bias,
                                                      int M, int N, int K, int actMode) {
  __shared__ bf16_t As[2][BM][LDP];
  __shared__ bf16_t Bs[2][BN][LDP];

  const int tid  = threadIdx.x;
  const int lane = tid & 31;
  const int wave = tid >> 5;
  const int wm   = wave & 3;   // 4 wave-rows of 32
  const int wn   = wave >> 2;  // 2 wave-cols of 64
  const int ln   = lane & 15;
  const int half = lane >> 4;

  const int row0 = blockIdx.y * BM;
  const int col0 = blockIdx.x * BN;

  v8f acc[2][4] = {};

  union Frag { v16bf v; bf16x8 h[2]; };

  const int kTiles = K / BK;

  // Issue one K-tile (A: 128x32, B: 128x32) as 4 async b128 loads per thread.
  auto issue = [&](int kt, int buf) {
    const int kk = kt * BK;
#pragma unroll
    for (int it = 0; it < 2; ++it) {
      int e = (tid + it * 256) * 8;
      int r = e >> 5;
      int c = e & 31;
      unsigned int la = (unsigned int)(size_t)&As[buf][r][c];
      async_load_b128(la, &A[(size_t)(row0 + r) * K + kk + c]);
    }
#pragma unroll
    for (int it = 0; it < 2; ++it) {
      int e = (tid + it * 256) * 8;
      int r = e >> 5;
      int c = e & 31;
      unsigned int la = (unsigned int)(size_t)&Bs[buf][r][c];
      async_load_b128(la, &BT[(size_t)(col0 + r) * K + kk + c]);
    }
  };

  // Fragments + 8 WMMAs on one staged buffer.
  auto compute = [&](int cur) {
    Frag af[2], bfq[4];
#pragma unroll
    for (int i = 0; i < 2; ++i) {
      int m = wm * 32 + i * 16 + ln;
      af[i].h[0] = *(const bf16x8*)&As[cur][m][half * 8];
      af[i].h[1] = *(const bf16x8*)&As[cur][m][16 + half * 8];
    }
#pragma unroll
    for (int j = 0; j < 4; ++j) {
      int n = wn * 64 + j * 16 + ln;
      bfq[j].h[0] = *(const bf16x8*)&Bs[cur][n][half * 8];
      bfq[j].h[1] = *(const bf16x8*)&Bs[cur][n][16 + half * 8];
    }
#pragma unroll
    for (int i = 0; i < 2; ++i)
#pragma unroll
      for (int j = 0; j < 4; ++j)
        acc[i][j] = __builtin_amdgcn_wmma_f32_16x16x32_bf16(
            false, af[i].v, false, bfq[j].v, (short)0, acc[i][j], false, false);
  };

  issue(0, 0);
  // Steady state: branch-free issue/wait/compute with double buffering.
  for (int kt = 0; kt < kTiles - 1; ++kt) {
    const int cur = kt & 1;
    issue(kt + 1, cur ^ 1);   // prefetch next tile into the other buffer
    wait_async_le4();         // only the 4 newest may remain -> tile kt done
    __syncthreads();          // all waves' async writes visible
    compute(cur);
    __syncthreads();          // buffer reusable for the next prefetch
  }
  // Peeled final tile.
  wait_async_0();
  __syncthreads();
  compute((kTiles - 1) & 1);

  // epilogue: D layout — lane L: n = L%16, VGPR r: m = r + 8*(L/16)
#pragma unroll
  for (int i = 0; i < 2; ++i) {
#pragma unroll
    for (int j = 0; j < 4; ++j) {
      int col = col0 + wn * 64 + j * 16 + ln;
      float bv = (actMode == 2) ? bias[col] : 0.f;
#pragma unroll
      for (int r = 0; r < 8; ++r) {
        int m = row0 + wm * 32 + i * 16 + r + 8 * half;
        float val = acc[i][j][r];
        if (actMode == 1)      val = 1.f / (1.f + expf(-val));
        else if (actMode == 2) val += bv;
        C[(size_t)m * N + col] = val;
      }
    }
  }
}

// ----------------------------- k projection --------------------------------
// One wave per row; 8-wide output reduced across lanes.

__global__ __launch_bounds__(256) void k_proj(const float* __restrict__ x,
                                              const float* __restrict__ Wk,  // K x 8
                                              float* __restrict__ kout,      // rows x 8
                                              int rows, int K) {
  int gwave = (int)((blockIdx.x * (long)blockDim.x + threadIdx.x) >> 5);
  int lane = threadIdx.x & 31;
  if (gwave >= rows) return;
  float acc[HEADS] = {};
  const float* xr = x + (size_t)gwave * K;
  for (int k = lane; k < K; k += 32) {
    float xv = xr[k];
    const float* wr = Wk + (size_t)k * HEADS;
#pragma unroll
    for (int p = 0; p < HEADS; ++p) acc[p] += xv * wr[p];
  }
#pragma unroll
  for (int p = 0; p < HEADS; ++p) {
    float v = acc[p];
    for (int off = 16; off > 0; off >>= 1) v += __shfl_down(v, off, 32);
    acc[p] = v;
  }
  if (lane == 0) {
    float* o = kout + (size_t)gwave * HEADS;
#pragma unroll
    for (int p = 0; p < HEADS; ++p) o[p] = acc[p];
  }
}

// --------------------------- gc / gcn reduction ----------------------------

__global__ __launch_bounds__(256) void gc_kernel(const float* __restrict__ kbuf,  // (b,n,8)
                                                 const float* __restrict__ v,     // (b,n,512)
                                                 float* __restrict__ gc,          // (b,8,64)
                                                 float* __restrict__ gcn) {       // (b,8)
  int b = blockIdx.x / HEADS;
  int p = blockIdx.x % HEADS;
  int t = threadIdx.x;
  int d = t & 63;
  int stripe = t >> 6;  // 0..3
  const float* kb = kbuf + (size_t)b * NTOK * HEADS;
  const float* vb = v + (size_t)b * NTOK * DIM + p * DH;
  float acc = 0.f, kn = 0.f;
  for (int i = stripe; i < NTOK; i += 4) {
    float ke = expf(kb[(size_t)i * HEADS + p]);
    acc += ke * vb[(size_t)i * DIM + d];
    if (d == 0) kn += ke;
  }
  __shared__ float sg[4][64];
  __shared__ float sk[4];
  sg[stripe][d] = acc;
  if (d == 0) sk[stripe] = kn;
  __syncthreads();
  if (t < 64) {
    gc[((size_t)b * HEADS + p) * DH + t] = sg[0][t] + sg[1][t] + sg[2][t] + sg[3][t];
    if (t == 0) gcn[b * HEADS + p] = sk[0] + sk[1] + sk[2] + sk[3];
  }
}

// ------------------- fused depthwise conv + combine ------------------------
// num = conv7x7(k*v) + gc ; den = conv7x7(k) + gcn + eps ; y = q*num/den (bf16)

static constexpr int TS   = 8;
static constexpr int HALO = TS + 6;  // 14

__global__ __launch_bounds__(256) void conv_combine(const float* __restrict__ kbuf,
                                                    const float* __restrict__ v,
                                                    const float* __restrict__ q,
                                                    const float* __restrict__ gc,
                                                    const float* __restrict__ gcn,
                                                    const float* __restrict__ wexp,
                                                    bf16_t* __restrict__ y) {
  __shared__ float vS[HALO][HALO][DH];
  __shared__ float kS[HALO][HALO];
  __shared__ float wS[49];

  const int tilesW = WW / TS;                      // 6
  const int tilesN = tilesW * (HH / TS);           // 36
  const int tileId = blockIdx.x % tilesN;
  const int bp = blockIdx.x / tilesN;
  const int b = bp / HEADS;
  const int p = bp % HEADS;
  const int th0 = (tileId / tilesW) * TS;
  const int tw0 = (tileId % tilesW) * TS;
  const int t = threadIdx.x;

  const float* kb = kbuf + (size_t)b * NTOK * HEADS + p;
  const float* vb = v + (size_t)b * NTOK * DIM + p * DH;

  for (int i = t; i < HALO * HALO * DH; i += 256) {
    int d = i & 63;
    int s = i >> 6;
    int ww = s % HALO;
    int hh = s / HALO;
    int gh = th0 + hh - 3, gw = tw0 + ww - 3;
    float val = 0.f;
    if (gh >= 0 && gh < HH && gw >= 0 && gw < WW)
      val = vb[(size_t)(gh * WW + gw) * DIM + d];
    vS[hh][ww][d] = val;
  }
  for (int i = t; i < HALO * HALO; i += 256) {
    int ww = i % HALO, hh = i / HALO;
    int gh = th0 + hh - 3, gw = tw0 + ww - 3;
    float val = 0.f;
    if (gh >= 0 && gh < HH && gw >= 0 && gw < WW)
      val = kb[(size_t)(gh * WW + gw) * HEADS];
    kS[hh][ww] = val;
  }
  if (t < 49) wS[t] = wexp[p * 49 + t];
  __syncthreads();

  const int d = t & 63;
  const float gcv  = gc[((size_t)b * HEADS + p) * DH + d];
  const float gcnv = gcn[b * HEADS + p];

  for (int s = t >> 6; s < TS * TS; s += 4) {
    int oh = s / TS, ow = s % TS;
    float num = gcv;
    float den = gcnv + EPSV;
#pragma unroll
    for (int dy = 0; dy < 7; ++dy)
#pragma unroll
      for (int dx = 0; dx < 7; ++dx) {
        float wk = wS[dy * 7 + dx] * kS[oh + dy][ow + dx];
        den += wk;
        num += wk * vS[oh + dy][ow + dx][d];
      }
    int gn = (th0 + oh) * WW + (tw0 + ow);
    size_t idx = (size_t)b * NTOK * DIM + (size_t)gn * DIM + p * DH + d;
    float outv = q[idx] * (num / den);
    y[idx] = (bf16_t)outv;
  }
}

// ------------------------------- launcher ----------------------------------

extern "C" void kernel_launch(void* const* d_in, const int* in_sizes, int n_in,
                              void* d_out, int out_size, void* d_ws, size_t ws_size,
                              hipStream_t stream) {
  const float* x     = (const float*)d_in[0];
  const float* Wq    = (const float*)d_in[1];
  const float* Wk    = (const float*)d_in[2];
  const float* Wv    = (const float*)d_in[3];
  const float* Wo    = (const float*)d_in[4];
  const float* bo    = (const float*)d_in[5];
  const float* w     = (const float*)d_in[6];
  const float* gamma = (const float*)d_in[7];
  const float* beta  = (const float*)d_in[8];
  (void)in_sizes; (void)n_in; (void)out_size; (void)ws_size;

  char* ws = (char*)d_ws;
  size_t off = 0;
  auto carve = [&](size_t bytes) -> void* {
    void* p = ws + off;
    off += (bytes + 255) & ~(size_t)255;
    return p;
  };

  bf16_t* xb   = (bf16_t*)carve((size_t)ROWS * DIM * 2);
  bf16_t* WqT  = (bf16_t*)carve((size_t)DIM * DIM * 2);
  bf16_t* WvT  = (bf16_t*)carve((size_t)DIM * DIM * 2);
  bf16_t* WoT  = (bf16_t*)carve((size_t)DIM * DIM * 2);
  float*  qbuf = (float*)carve((size_t)ROWS * DIM * 4);
  float*  vbuf = (float*)carve((size_t)ROWS * DIM * 4);
  float*  kbuf = (float*)carve((size_t)ROWS * HEADS * 4);
  float*  gcB  = (float*)carve((size_t)BATCH * HEADS * DH * 4);
  float*  gcnB = (float*)carve((size_t)BATCH * HEADS * 4);
  float*  weB  = (float*)carve((size_t)HEADS * 49 * 4);
  bf16_t* yb   = (bf16_t*)carve((size_t)ROWS * DIM * 2);

  // 1. conversions
  {
    long total = (long)ROWS * DIM;
    int blocks = (int)((total + 255) / 256);
    if (blocks > 8192) blocks = 8192;
    convert_bf16<<<blocks, 256, 0, stream>>>(x, xb, total);
  }
  {
    int blocks = (DIM * DIM + 255) / 256;
    transpose_to_bf16<<<blocks, 256, 0, stream>>>(Wq, WqT, DIM, DIM);
    transpose_to_bf16<<<blocks, 256, 0, stream>>>(Wv, WvT, DIM, DIM);
    transpose_to_bf16<<<blocks, 256, 0, stream>>>(Wo, WoT, DIM, DIM);
  }
  wexp_kernel<<<1, 128, 0, stream>>>(w, gamma, beta, weB);

  // 2. projections
  dim3 gg(DIM / BN, ROWS / BM);
  gemm_bf16_wmma<<<gg, 256, 0, stream>>>(xb, WqT, qbuf, nullptr, ROWS, DIM, DIM, 1);
  gemm_bf16_wmma<<<gg, 256, 0, stream>>>(xb, WvT, vbuf, nullptr, ROWS, DIM, DIM, 0);
  k_proj<<<(ROWS * 32) / 256, 256, 0, stream>>>(x, Wk, kbuf, ROWS, DIM);

  // 3. global context
  gc_kernel<<<BATCH * HEADS, 256, 0, stream>>>(kbuf, vbuf, gcB, gcnB);

  // 4. fused depthwise conv + combine -> y (bf16)
  conv_combine<<<BATCH * HEADS * 36, 256, 0, stream>>>(kbuf, vbuf, qbuf, gcB, gcnB,
                                                       weB, yb);

  // 5. output projection -> d_out (f32) with bias
  gemm_bf16_wmma<<<gg, 256, 0, stream>>>(yb, WoT, (float*)d_out, bo, ROWS, DIM, DIM, 2);
}
